// JointLearningModel_5755256177375
// MI455X (gfx1250) — matrix operand order repaired and checked by reading, not compiled, not run on hardware
//
#include <hip/hip_runtime.h>
#include <hip/hip_bf16.h>

typedef __attribute__((ext_vector_type(16))) _Float16 v16h;
typedef __attribute__((ext_vector_type(2)))  _Float16 v2h;
typedef __attribute__((ext_vector_type(8)))  float    v8f;

#define Bdim 8
#define Ldim 512
#define Hdim 768
#define Hh   384   // H/2
#define Hq   192   // H/4
#define Mm   383
#define Nn   384
#define NEGV (-10000.0f)

// ---------------------------------------------------------------------------
// 1) Gather: all_reps[0] = dummy; all_reps[1+m] = start + end + speaker
// ---------------------------------------------------------------------------
__global__ void gather_kernel(const float* __restrict__ seq,
                              const float* __restrict__ spk_emb,
                              const float* __restrict__ dummy,
                              const int* __restrict__ seg,
                              const int* __restrict__ mstart,
                              const int* __restrict__ mend,
                              const int* __restrict__ spk_ids,
                              float* __restrict__ all_reps) {
  int idx = blockIdx.x * blockDim.x + threadIdx.x;
  if (idx >= Nn * Hdim) return;
  int n = idx / Hdim, k = idx % Hdim;
  float v;
  if (n == 0) {
    v = dummy[k];
  } else {
    int m = n - 1;
    int s = seg[m], st = mstart[m], en = mend[m];
    int sid = spk_ids[s * Ldim + st];
    v = seq[(size_t)(s * Ldim + st) * Hdim + k] +
        seq[(size_t)(s * Ldim + en) * Hdim + k] +
        spk_emb[(size_t)sid * Hdim + k];
  }
  all_reps[idx] = v;
}

// ---------------------------------------------------------------------------
// 2) Generic fp32 linear: Y[n,o] = act( X[n,:]·W[o, woff:woff+K] + b[o] )
//    (used for the small GEMMs: A/B projections, ms-MLP, char head)
// ---------------------------------------------------------------------------
__global__ void lin_kernel(const float* __restrict__ X, const float* __restrict__ W,
                           const float* __restrict__ bias, float* __restrict__ Y,
                           int N, int K, int O, int wld, int woff, int dorelu) {
  int idx = blockIdx.x * blockDim.x + threadIdx.x;
  if (idx >= N * O) return;
  int n = idx / O, o = idx % O;
  const float* x = X + (size_t)n * K;
  const float* w = W + (size_t)o * wld + woff;
  float acc = bias ? bias[o] : 0.0f;
  for (int k = 0; k < K; ++k) acc = fmaf(x[k], w[k], acc);
  if (dorelu) acc = fmaxf(acc, 0.0f);
  Y[idx] = acc;
}

// ---------------------------------------------------------------------------
// 3) Pack W_pair2 (Hh x H, fp32, h-major) -> f16 pairs, K-pair-major:
//    W2pk[(k/2)*Hh + h] = { W2[h,k], W2[h,k+1] }   (matches WMMA B layout)
// ---------------------------------------------------------------------------
__global__ void pack_w2_kernel(const float* __restrict__ W2, _Float16* __restrict__ W2pk) {
  int idx = blockIdx.x * blockDim.x + threadIdx.x;
  if (idx >= (Hdim / 2) * Hh) return;
  int prow = idx / Hh, h = idx % Hh;
  int k = prow * 2;
  v2h p;
  p.x = (_Float16)W2[(size_t)h * Hdim + k];
  p.y = (_Float16)W2[(size_t)h * Hdim + k + 1];
  ((v2h*)W2pk)[idx] = p;
}

// ---------------------------------------------------------------------------
// 4) The 87-GFLOP pairwise MLP, on WMMA f16->f32.
//    grid = (24 j-tiles, 384 i). Block = 256 (8 waves); wave w owns hidden
//    tiles {w, w+8, w+16}. For each 32-wide k-step:
//      A-frag = relu(A[j,k] + B[i,k] + b1[k]) packed to f16 (ISA A layout)
//      B-frag = prepacked W2 f16 pairs      (ISA B layout)
//    72 v_wmma_f32_16x16x32_f16 per wave; epilogue fuses +b2, relu, ·w3 and
//    a deterministic fixed-order LDS combine (no float atomics).
// ---------------------------------------------------------------------------
__global__ void __launch_bounds__(256)
pair_kernel(const float* __restrict__ Amat, const float* __restrict__ Bmat,
            const float* __restrict__ b1, const _Float16* __restrict__ W2pk,
            const float* __restrict__ b2, const float* __restrict__ w3,
            const float* __restrict__ b3, float* __restrict__ pair_out) {
  __shared__ float sC[Hdim];       // B[i,:] + b1
  __shared__ float sPart[24 * 16]; // per-hidden-tile partial j-scores

  const int i     = blockIdx.y;
  const int jbase = blockIdx.x * 16;
  const int tid   = threadIdx.x;

  for (int k = tid; k < Hdim; k += 256) sC[k] = Bmat[(size_t)i * Hdim + k] + b1[k];
  __syncthreads();

  const int  lane = tid & 31;
  const int  w    = tid >> 5;
  const int  n    = lane & 15;
  const bool hi   = lane >= 16;
  const int  selA = hi ? 8 : 0;

  const float* Arow = Amat + (size_t)(jbase + n) * Hdim;
  const v2h*   W2v  = (const v2h*)W2pk;

  v8f acc[3] = {v8f{}, v8f{}, v8f{}};

  for (int kb = 0; kb < Hdim; kb += 32) {
    // Build A fragment: 16x32 f16, VGPR v holds K = {base + 2(v&3), +1},
    // base = ((v&4)?16:0) + (hi?8:0)
    v16h a;
#pragma unroll
    for (int v = 0; v < 8; ++v) {
      int k = kb + ((v & 4) ? 16 : 0) + selA + 2 * (v & 3);
      float a0 = fmaxf(Arow[k] + sC[k], 0.0f);
      float a1 = fmaxf(Arow[k + 1] + sC[k + 1], 0.0f);
      a[2 * v]     = (_Float16)a0;
      a[2 * v + 1] = (_Float16)a1;
    }
    const int prow = (kb >> 1) + (selA >> 1); // K-pair row for this lane half
#pragma unroll
    for (int t = 0; t < 3; ++t) {
      const int hb = (w + t * 8) * 16;
      v16h b;
#pragma unroll
      for (int v = 0; v < 8; ++v) {
        v2h p = W2v[(size_t)(prow + v + (v & 4)) * Hh + hb + n];
        b[2 * v]     = p.x;
        b[2 * v + 1] = p.y;
      }
      acc[t] = __builtin_amdgcn_wmma_f32_16x16x32_f16(
          false, a, false, b, (short)0, acc[t], false, false);
    }
  }

  // Epilogue: relu(acc + b2[h]) * w3[h], reduce over the 16 h-lanes.
#pragma unroll
  for (int t = 0; t < 3; ++t) {
    const int ht = w + t * 8;
    const int h  = ht * 16 + n;
    const float bb = b2[h];
    const float ww = w3[h];
#pragma unroll
    for (int v = 0; v < 8; ++v) {
      float val = fmaxf(acc[t][v] + bb, 0.0f) * ww;
      val += __shfl_xor(val, 1);
      val += __shfl_xor(val, 2);
      val += __shfl_xor(val, 4);
      val += __shfl_xor(val, 8);
      if (n == 0) sPart[ht * 16 + v + selA] = val; // row j = v + 8*hi
    }
  }
  __syncthreads();

  if (tid < 16) {
    float s = 0.0f;
    for (int ht = 0; ht < 24; ++ht) s += sPart[ht * 16 + tid];
    pair_out[(size_t)i * Nn + jbase + tid] = s + b3[0];
  }
}

// ---------------------------------------------------------------------------
// 5) Per-row log-softmax term: row i (1..383):
//    nll_i = lse_j(pair[i,j]+ms[i]+ms[j]+mask) - score(i, first[i-1])
// ---------------------------------------------------------------------------
__global__ void row_kernel(const float* __restrict__ pair, const float* __restrict__ ms,
                           const int* __restrict__ link_first, float* __restrict__ row_nll) {
  const int i = blockIdx.x + 1;
  const int tid = threadIdx.x; // 128
  __shared__ float red[128];
  const float msi = ms[i];
  float vals[3];
  float lmax = -3.4e38f;
#pragma unroll
  for (int t = 0; t < 3; ++t) {
    int j = tid + t * 128;
    float v = pair[(size_t)i * Nn + j] + msi + ms[j] + (j >= i ? NEGV : 0.0f);
    vals[t] = v;
    lmax = fmaxf(lmax, v);
  }
  red[tid] = lmax;
  __syncthreads();
  for (int s = 64; s > 0; s >>= 1) {
    if (tid < s) red[tid] = fmaxf(red[tid], red[tid + s]);
    __syncthreads();
  }
  const float mx = red[0];
  __syncthreads();
  float se = 0.0f;
#pragma unroll
  for (int t = 0; t < 3; ++t) se += expf(vals[t] - mx);
  red[tid] = se;
  __syncthreads();
  for (int s = 64; s > 0; s >>= 1) {
    if (tid < s) red[tid] += red[tid + s];
    __syncthreads();
  }
  if (tid == 0) {
    int jf = link_first[i - 1];
    float sc = pair[(size_t)i * Nn + jf] + msi + ms[jf] + (jf >= i ? NEGV : 0.0f);
    row_nll[i - 1] = (mx + logf(red[0])) - sc;
  }
}

// ---------------------------------------------------------------------------
// 6) Char-head cross entropy per mention
// ---------------------------------------------------------------------------
__global__ void ce_kernel(const float* __restrict__ logits, const int* __restrict__ label,
                          float* __restrict__ ce_arr) {
  int m = blockIdx.x * blockDim.x + threadIdx.x;
  if (m >= Mm) return;
  const float* l = logits + (size_t)m * 18;
  float mx = l[0];
  for (int k = 1; k < 18; ++k) mx = fmaxf(mx, l[k]);
  float se = 0.0f;
  for (int k = 0; k < 18; ++k) se += expf(l[k] - mx);
  ce_arr[m] = (mx + logf(se)) - l[label[m]];
}

// ---------------------------------------------------------------------------
// 7) Deterministic final scalar sum
// ---------------------------------------------------------------------------
__global__ void final_kernel(const float* __restrict__ row_nll,
                             const float* __restrict__ ce_arr, float* __restrict__ out) {
  if (threadIdx.x == 0 && blockIdx.x == 0) {
    float s = 0.0f;
    for (int i = 0; i < Mm; ++i) s += row_nll[i];
    for (int i = 0; i < Mm; ++i) s += ce_arr[i];
    out[0] = s;
  }
}

// ---------------------------------------------------------------------------
extern "C" void kernel_launch(void* const* d_in, const int* in_sizes, int n_in,
                              void* d_out, int out_size, void* d_ws, size_t ws_size,
                              hipStream_t stream) {
  (void)in_sizes; (void)n_in; (void)out_size; (void)ws_size;

  const float* seq      = (const float*)d_in[0];
  const float* spk_emb  = (const float*)d_in[1];
  const float* dummy    = (const float*)d_in[2];
  const float* W_pair1  = (const float*)d_in[3];
  const float* b_pair1  = (const float*)d_in[4];
  const float* W_pair2  = (const float*)d_in[5];
  const float* b_pair2  = (const float*)d_in[6];
  const float* W_pair3  = (const float*)d_in[7];
  const float* b_pair3  = (const float*)d_in[8];
  const float* W_m1     = (const float*)d_in[9];
  const float* b_m1     = (const float*)d_in[10];
  const float* W_m2     = (const float*)d_in[11];
  const float* b_m2     = (const float*)d_in[12];
  const float* W_m3     = (const float*)d_in[13];
  const float* b_m3     = (const float*)d_in[14];
  const float* W_c1     = (const float*)d_in[15];
  const float* b_c1     = (const float*)d_in[16];
  const float* W_c2     = (const float*)d_in[17];
  const float* b_c2     = (const float*)d_in[18];
  const int* seg        = (const int*)d_in[19];
  const int* mstart     = (const int*)d_in[20];
  const int* mend       = (const int*)d_in[21];
  const int* spk_ids    = (const int*)d_in[22];
  const int* link_first = (const int*)d_in[23];
  const int* label      = (const int*)d_in[25];

  // workspace layout (float slots)
  float* ws        = (float*)d_ws;
  float* all_reps  = ws;                         // 384*768
  float* Amat      = all_reps + Nn * Hdim;       // 384*768
  float* Bmat      = Amat + Nn * Hdim;           // 384*768
  _Float16* W2pk   = (_Float16*)(Bmat + Nn * Hdim); // 768*384 halves = 147456 float slots
  float* pair      = Bmat + Nn * Hdim + (Hdim * Hh) / 2;
  float* ms1       = pair + Nn * Nn;             // 384*384
  float* ms2       = ms1 + Nn * Hh;              // 384*192
  float* msv       = ms2 + Nn * Hq;              // 384
  float* c1        = msv + Nn;                   // 383*384
  float* logits    = c1 + Mm * Hh;               // 383*18
  float* ce_arr    = logits + Mm * 18;           // 383
  float* row_nll   = ce_arr + Mm;                // 383

  const int T = 256;

  gather_kernel<<<(Nn * Hdim + T - 1) / T, T, 0, stream>>>(
      seq, spk_emb, dummy, seg, mstart, mend, spk_ids, all_reps);

  // A = all_reps @ Wa^T ; Bm = all_reps @ Wb^T   (Wa|Wb = W_pair1 split)
  lin_kernel<<<(Nn * Hdim + T - 1) / T, T, 0, stream>>>(
      all_reps, W_pair1, nullptr, Amat, Nn, Hdim, Hdim, 2 * Hdim, 0, 0);
  lin_kernel<<<(Nn * Hdim + T - 1) / T, T, 0, stream>>>(
      all_reps, W_pair1, nullptr, Bmat, Nn, Hdim, Hdim, 2 * Hdim, Hdim, 0);

  pack_w2_kernel<<<((Hdim / 2) * Hh + T - 1) / T, T, 0, stream>>>(W_pair2, W2pk);

  pair_kernel<<<dim3(Nn / 16, Nn), T, 0, stream>>>(
      Amat, Bmat, b_pair1, W2pk, b_pair2, W_pair3, b_pair3, pair);

  // ms-MLP over all_reps
  lin_kernel<<<(Nn * Hh + T - 1) / T, T, 0, stream>>>(
      all_reps, W_m1, b_m1, ms1, Nn, Hdim, Hh, Hdim, 0, 1);
  lin_kernel<<<(Nn * Hq + T - 1) / T, T, 0, stream>>>(
      ms1, W_m2, b_m2, ms2, Nn, Hh, Hq, Hh, 0, 1);
  lin_kernel<<<(Nn + T - 1) / T, T, 0, stream>>>(
      ms2, W_m3, b_m3, msv, Nn, Hq, 1, Hq, 0, 0);

  // char head over mention_reps (rows 1..383 of all_reps)
  lin_kernel<<<(Mm * Hh + T - 1) / T, T, 0, stream>>>(
      all_reps + Hdim, W_c1, b_c1, c1, Mm, Hdim, Hh, Hdim, 0, 1);
  lin_kernel<<<(Mm * 18 + T - 1) / T, T, 0, stream>>>(
      c1, W_c2, b_c2, logits, Mm, Hh, 18, Hh, 0, 0);
  ce_kernel<<<(Mm + T - 1) / T, T, 0, stream>>>(logits, label, ce_arr);

  row_kernel<<<Mm, 128, 0, stream>>>(pair, msv, link_first, row_nll);
  final_kernel<<<1, 32, 0, stream>>>(row_nll, ce_arr, (float*)d_out);
}
